// SwinTransformer_48352741818788
// MI455X (gfx1250) — matrix operand order repaired and checked by reading, not compiled, not run on hardware
//
#include <hip/hip_runtime.h>
#include <hip/hip_bf16.h>
#include <math.h>

// ---------------------------------------------------------------------------
// Swin block, bf16 WMMA pipeline for gfx1250 (wave32).
//   B=32 H=W=56 C=384 HEADS=12 hd=32 WIN=7 N=49 SHIFT=3 MLP_HID=1536
//   M (tokens) = 32*3136 = 100352  (= 784 * 128, exact tiling)
// GEMMs use double-buffered GLOBAL_LOAD_ASYNC_TO_LDS staging (ASYNCcnt) so
// WMMA issue overlaps the global->LDS traffic.
// ---------------------------------------------------------------------------

typedef __attribute__((ext_vector_type(16))) __bf16 bf16x16;
typedef __attribute__((ext_vector_type(8)))  __bf16 bf16x8;
typedef __attribute__((ext_vector_type(8)))  float  f32x8;

#define MTOK   100352
#define CDIM   384
#define QKVDIM 1152
#define HIDDIM 1536
#define NHEAD  12
#define HD     32
#define NWIN   2048      // 32 batches * 64 windows
#define NTOK   49
#define QSCALE 0.17677669529663689f   // 32^-0.5

__device__ __forceinline__ f32x8 wmma_bf16(bf16x16 a, bf16x16 b, f32x8 c) {
  return __builtin_amdgcn_wmma_f32_16x16x32_bf16(false, a, false, b, (short)0, c,
                                                 false, false);
}

__device__ __forceinline__ bf16x16 concat8(bf16x8 lo, bf16x8 hi) {
  return __builtin_shufflevector(lo, hi, 0, 1, 2, 3, 4, 5, 6, 7,
                                 8, 9, 10, 11, 12, 13, 14, 15);
}

// A-fragment (16x32 bf16): lane holds row M=lane%16; lanes<16 carry K
// {k0..k0+7, k0+16..k0+23}, lanes>=16 carry {k0+8.., k0+24..}. Source LDS
// layout [M][lda] row-major -> two contiguous 16B loads per lane.
__device__ __forceinline__ bf16x16 load_a_frag(const __bf16* s, int m0, int k0,
                                               int lda) {
  const int lane = threadIdx.x & 31;
  const __bf16* p = s + (m0 + (lane & 15)) * lda + k0 + ((lane >> 4) << 3);
  bf16x8 lo = *(const bf16x8*)(p);
  bf16x8 hi = *(const bf16x8*)(p + 16);
  return concat8(lo, hi);
}

// B-fragment (32x16 bf16): lane holds column N=lane%16; lanes<16 carry
// K {k0..k0+7, k0+16..k0+23}, lanes>=16 carry {k0+8.., k0+24..}. Source LDS layout [N][ldb]
// (k-contiguous per output column) -> two contiguous 16B loads per lane.
__device__ __forceinline__ bf16x16 load_b_frag(const __bf16* s, int n0, int k0,
                                               int ldb) {
  const int lane = threadIdx.x & 31;
  const __bf16* p = s + (n0 + (lane & 15)) * ldb + k0 + ((lane >> 4) << 3);
  bf16x8 lo = *(const bf16x8*)(p);
  bf16x8 hi = *(const bf16x8*)(p + 16);
  return concat8(lo, hi);
}

// Async global->LDS 16-byte copy per lane (VGLOBAL encoding, ASYNCcnt).
// LDS byte address = low 32 bits of the generic pointer to a __shared__
// object (generic LDS addresses are {shared_aperture_hi, lds_offset}).
__device__ __forceinline__ void async_copy16(const __bf16* g, __bf16* l) {
  unsigned lds = (unsigned)(unsigned long)l;
  unsigned long ga = (unsigned long)g;
  asm volatile("global_load_async_to_lds_b128 %0, %1, off"
               :: "v"(lds), "v"(ga) : "memory");
}

__device__ __forceinline__ void wait_async_all() {
  asm volatile("s_wait_asynccnt 0x0" ::: "memory");
}

// ---------------------------------------------------------------------------
// f32 -> bf16 weight conversion
// ---------------------------------------------------------------------------
__global__ void cvt_kernel(const float* __restrict__ src,
                           __bf16* __restrict__ dst, int n) {
  int i = blockIdx.x * blockDim.x + threadIdx.x;
  if (i < n) dst[i] = (__bf16)src[i];
}

// ---------------------------------------------------------------------------
// LayerNorm (+ optional cyclic shift & window partition gather), bf16 out.
// shifted=1: block row r is window-major (w*49+n) gathered from the rolled
// image; shifted=0: identity row mapping (used for LN2 on xres).
// ---------------------------------------------------------------------------
__global__ __launch_bounds__(384) void ln_kernel(const float* __restrict__ src,
                                                 const float* __restrict__ g,
                                                 const float* __restrict__ beta,
                                                 __bf16* __restrict__ out,
                                                 int shifted) {
  __shared__ float wsum[12], wsq[12];
  __shared__ float s_mean, s_rstd;
  const long row = blockIdx.x;
  const int c = threadIdx.x;
  long srow;
  if (shifted) {
    int w = (int)(row / NTOK), n = (int)(row % NTOK);
    int b = w >> 6, wy = (w >> 3) & 7, wx = w & 7;
    int sh = (wy * 7 + n / 7 + 3) % 56;   // roll(-SHIFT) gather
    int sw = (wx * 7 + n % 7 + 3) % 56;
    srow = (long)b * 3136 + sh * 56 + sw;
  } else {
    srow = row;
  }
  float v = src[srow * CDIM + c];
  float s = v, s2 = v * v;
  for (int o = 16; o; o >>= 1) {
    s  += __shfl_down(s,  o, 32);
    s2 += __shfl_down(s2, o, 32);
  }
  if ((c & 31) == 0) { wsum[c >> 5] = s; wsq[c >> 5] = s2; }
  __syncthreads();
  if (c == 0) {
    float ts = 0.f, tq = 0.f;
    for (int i = 0; i < 12; ++i) { ts += wsum[i]; tq += wsq[i]; }
    float m = ts * (1.0f / CDIM);
    float var = tq * (1.0f / CDIM) - m * m;
    s_mean = m;
    s_rstd = rsqrtf(var + 1e-5f);
  }
  __syncthreads();
  float r = (v - s_mean) * s_rstd * g[c] + beta[c];
  out[row * CDIM + c] = (__bf16)r;
}

// ---------------------------------------------------------------------------
// WMMA GEMM: OUT[M,N] = epilogue(A[M,K] @ W[N,K]^T + bias)
// Tile 128x128x32, 256 threads = 8 waves, each wave 32x64 (2x4 WMMA tiles).
// Double-buffered async global->LDS staging: stage tile k+1 while computing
// tile k; one s_wait_asynccnt+barrier per K-step.
// EPI: 0=QKV(bf16 out, scale q cols), 1=PROJ(window-reverse scatter+residual,
//      f32 out), 2=FC1(GELU, bf16 out), 3=FC2(+residual, f32 out)
// ---------------------------------------------------------------------------
#define BM 128
#define BN 128
#define BK 32

__device__ __forceinline__ void stage_async(const __bf16* __restrict__ A,
                                            const __bf16* __restrict__ W,
                                            long m_base, int n_base, long Kdim,
                                            int k0, __bf16* as, __bf16* bs,
                                            int tid) {
  const int r = tid >> 2;            // 0..63
  const int q = (tid & 3) * 8;       // 0/8/16/24 (bf16 elems) = 16B chunks
  async_copy16(A + (m_base + r) * Kdim + k0 + q, as + r * BK + q);
  async_copy16(A + (m_base + r + 64) * Kdim + k0 + q, as + (r + 64) * BK + q);
  async_copy16(W + (long)(n_base + r) * Kdim + k0 + q, bs + r * BK + q);
  async_copy16(W + (long)(n_base + r + 64) * Kdim + k0 + q, bs + (r + 64) * BK + q);
}

template <int EPI>
__global__ __launch_bounds__(256) void gemm_kernel(
    const __bf16* __restrict__ A, const __bf16* __restrict__ W,
    const float* __restrict__ bias, __bf16* __restrict__ outb,
    float* __restrict__ outf, const float* __restrict__ add, int Ndim,
    int Kdim) {
  __shared__ __align__(16) __bf16 As[2][BM * BK];   // [M][K], double buffered
  __shared__ __align__(16) __bf16 Bs[2][BN * BK];   // [N][K], double buffered

  const int tid = threadIdx.x;
  const int lane = tid & 31;
  const int wave = tid >> 5;
  const int wm = wave & 3;     // M 32-slab
  const int wn = wave >> 2;    // N 64-slab
  const long m_base = (long)blockIdx.x * BM;
  const int n_base = blockIdx.y * BN;

  f32x8 acc[2][4] = {};

  const int ksteps = Kdim / BK;
  stage_async(A, W, m_base, n_base, Kdim, 0, As[0], Bs[0], tid);
  wait_async_all();
  __syncthreads();

  for (int kt = 0; kt < ksteps; ++kt) {
    const int cur = kt & 1;
    if (kt + 1 < ksteps)
      stage_async(A, W, m_base, n_base, Kdim, (kt + 1) * BK, As[cur ^ 1],
                  Bs[cur ^ 1], tid);

    bf16x16 a0 = load_a_frag(As[cur], wm * 32, 0, BK);
    bf16x16 a1 = load_a_frag(As[cur], wm * 32 + 16, 0, BK);
#pragma unroll
    for (int j = 0; j < 4; ++j) {
      bf16x16 b = load_b_frag(Bs[cur], wn * 64 + j * 16, 0, BK);
      acc[0][j] = wmma_bf16(a0, b, acc[0][j]);
      acc[1][j] = wmma_bf16(a1, b, acc[1][j]);
    }

    if (kt + 1 < ksteps) {
      wait_async_all();     // next buffer landed in LDS
      __syncthreads();      // all waves done reading current buffer
    }
  }

  // Epilogue. C/D layout: VGPR r -> M = r + 8*(lane>=16); N = lane%16.
  const int mh = (lane >> 4) << 3;
  const int ncol = lane & 15;
#pragma unroll
  for (int i = 0; i < 2; ++i) {
#pragma unroll
    for (int j = 0; j < 4; ++j) {
#pragma unroll
      for (int r = 0; r < 8; ++r) {
        const long gm = m_base + wm * 32 + i * 16 + r + mh;
        const int gn = n_base + wn * 64 + j * 16 + ncol;
        float v = acc[i][j][r] + bias[gn];
        if (EPI == 0) {          // QKV: scale Q columns
          float sc = (gn < CDIM) ? QSCALE : 1.0f;
          outb[gm * QKVDIM + gn] = (__bf16)(v * sc);
        } else if (EPI == 1) {   // PROJ: window-reverse + unshift + residual
          int w = (int)(gm / NTOK), n = (int)(gm % NTOK);
          int b = w >> 6, wy = (w >> 3) & 7, wx = w & 7;
          int dh = (wy * 7 + n / 7 + 3) % 56;   // roll(+SHIFT) scatter
          int dw = (wx * 7 + n % 7 + 3) % 56;
          long dest = (long)b * 3136 + dh * 56 + dw;
          outf[dest * CDIM + gn] = add[dest * CDIM + gn] + v;
        } else if (EPI == 2) {   // FC1: exact GELU
          float gl = 0.5f * v * (1.0f + erff(v * 0.70710678118654752f));
          outb[gm * (long)HIDDIM + gn] = (__bf16)gl;
        } else {                 // FC2: + residual -> final output
          outf[gm * CDIM + gn] = v + add[gm * CDIM + gn];
        }
      }
    }
  }
}

// ---------------------------------------------------------------------------
// Attention: one block per (window, head). 128 threads = 4 waves.
// S = (Q*scale) K^T (+bias,+mask) -> softmax -> O = P V.
// 49 padded to 64; hd=32 => QK^T is a single WMMA K-step per tile.
// ---------------------------------------------------------------------------
__global__ __launch_bounds__(128) void attn_kernel(
    const __bf16* __restrict__ qkv, const float* __restrict__ rel_bias,
    __bf16* __restrict__ ao) {
  __shared__ __align__(16) __bf16 Qs[64 * 32];   // [tok][hd]
  __shared__ __align__(16) __bf16 Ks[64 * 32];   // [tok][hd]
  __shared__ __align__(16) __bf16 Vt[32 * 64];   // [hd][tok] (transposed)
  __shared__ float Ss[64 * 64];
  __shared__ __align__(16) __bf16 Ps[64 * 64];

  const int w = blockIdx.x, head = blockIdx.y;
  const int tid = threadIdx.x, lane = tid & 31, wave = tid >> 5;
  const int wy = (w >> 3) & 7, wx = w & 7;

  for (int t = tid; t < NTOK * HD; t += 128) {
    int tok = t >> 5, c = t & 31;
    const __bf16* base = qkv + ((long)w * NTOK + tok) * QKVDIM + head * HD + c;
    Qs[tok * 32 + c] = base[0];
    Ks[tok * 32 + c] = base[CDIM];
    Vt[c * 64 + tok] = base[2 * CDIM];
  }
  for (int t = tid; t < (64 - NTOK) * HD; t += 128) {   // zero padding rows
    int tok = NTOK + (t >> 5), c = t & 31;
    Qs[tok * 32 + c] = (__bf16)0.0f;
    Ks[tok * 32 + c] = (__bf16)0.0f;
    Vt[c * 64 + tok] = (__bf16)0.0f;
  }
  __syncthreads();

  // S = Q @ K^T   (each wave owns a 16-row slab; 4 WMMAs)
  {
    const int m0 = wave * 16;
    bf16x16 a = load_a_frag(Qs, m0, 0, 32);
    for (int nt = 0; nt < 4; ++nt) {
      f32x8 d = {};
      bf16x16 b = load_b_frag(Ks, nt * 16, 0, 32);
      d = wmma_bf16(a, b, d);
#pragma unroll
      for (int r = 0; r < 8; ++r)
        Ss[(m0 + r + ((lane >> 4) << 3)) * 64 + nt * 16 + (lane & 15)] = d[r];
    }
  }
  __syncthreads();

  // Softmax with fused relative-position bias and shift mask (one thread/row)
  if (tid < 64) {
    const int i = tid;
    if (i < NTOK) {
      const int ih = i / 7, iw = i % 7;
      const int rr = wy * 7 + ih, rc = wx * 7 + iw;
      const int li = (rr < 49 ? 0 : (rr < 53 ? 1 : 2)) * 3 +
                     (rc < 49 ? 0 : (rc < 53 ? 1 : 2));
      float mx = -1e30f;
      for (int j = 0; j < NTOK; ++j) {
        int jh = j / 7, jw = j % 7;
        int jr = wy * 7 + jh, jc = wx * 7 + jw;
        int lj = (jr < 49 ? 0 : (jr < 53 ? 1 : 2)) * 3 +
                 (jc < 49 ? 0 : (jc < 53 ? 1 : 2));
        float s = Ss[i * 64 + j] +
                  rel_bias[((ih - jh + 6) * 13 + (iw - jw + 6)) * NHEAD + head];
        if (li != lj) s -= 100.0f;
        Ss[i * 64 + j] = s;
        mx = fmaxf(mx, s);
      }
      float sum = 0.f;
      for (int j = 0; j < NTOK; ++j) {
        float e = expf(Ss[i * 64 + j] - mx);
        Ss[i * 64 + j] = e;
        sum += e;
      }
      const float inv = 1.0f / sum;
      for (int j = 0; j < NTOK; ++j) Ps[i * 64 + j] = (__bf16)(Ss[i * 64 + j] * inv);
      for (int j = NTOK; j < 64; ++j) Ps[i * 64 + j] = (__bf16)0.0f;
    } else {
      for (int j = 0; j < 64; ++j) Ps[i * 64 + j] = (__bf16)0.0f;
    }
  }
  __syncthreads();

  // O = P @ V   (K=64 -> two WMMA steps; 2 column tiles of 16)
  {
    const int m0 = wave * 16;
    bf16x16 a0 = load_a_frag(Ps, m0, 0, 64);
    bf16x16 a1 = load_a_frag(Ps, m0, 32, 64);
    for (int j = 0; j < 2; ++j) {
      f32x8 acc = {};
      bf16x16 b0 = load_b_frag(Vt, j * 16, 0, 64);
      bf16x16 b1 = load_b_frag(Vt, j * 16, 32, 64);
      acc = wmma_bf16(a0, b0, acc);
      acc = wmma_bf16(a1, b1, acc);
#pragma unroll
      for (int r = 0; r < 8; ++r) {
        int m = m0 + r + ((lane >> 4) << 3);
        if (m < NTOK) {
          long orow = (long)w * NTOK + m;
          ao[orow * CDIM + head * HD + j * 16 + (lane & 15)] = (__bf16)acc[r];
        }
      }
    }
  }
}

// ---------------------------------------------------------------------------
// Host launch. Workspace layout (bytes):
//   hbuf   bf16 [M][384]       @ 0           (77,070,336)   -> reused as h2buf
//   qkvbuf bf16 [M][1152]      @ 77,070,336  (231,211,008)  -\ reused together
//   aobuf  bf16 [M][384]       @ 308,281,344 (77,070,336)   -/ as hidden[M][1536]
//   xres   f32  [M][384]       @ 385,351,680 (154,140,672)
//   bf16 weights               @ 539,492,352 (~3.5 MB)       total ~543 MB
// ---------------------------------------------------------------------------
extern "C" void kernel_launch(void* const* d_in, const int* in_sizes, int n_in,
                              void* d_out, int out_size, void* d_ws,
                              size_t ws_size, hipStream_t stream) {
  const float* x      = (const float*)d_in[0];
  const float* g1     = (const float*)d_in[1];
  const float* b1     = (const float*)d_in[2];
  const float* qkv_w  = (const float*)d_in[3];
  const float* qkv_b  = (const float*)d_in[4];
  const float* relb   = (const float*)d_in[5];
  const float* proj_w = (const float*)d_in[6];
  const float* proj_b = (const float*)d_in[7];
  const float* g2     = (const float*)d_in[8];
  const float* b2     = (const float*)d_in[9];
  const float* fc1_w  = (const float*)d_in[10];
  const float* fc1_b  = (const float*)d_in[11];
  const float* fc2_w  = (const float*)d_in[12];
  const float* fc2_b  = (const float*)d_in[13];

  char* ws = (char*)d_ws;
  __bf16* hbuf   = (__bf16*)(ws);
  __bf16* qkvbuf = (__bf16*)(ws + 77070336UL);
  __bf16* aobuf  = (__bf16*)(ws + 308281344UL);
  float*  xres   = (float*)(ws + 385351680UL);
  __bf16* wqkv   = (__bf16*)(ws + 539492352UL);
  __bf16* wproj  = (__bf16*)(ws + 539492352UL + 884736UL);
  __bf16* wfc1   = (__bf16*)(ws + 539492352UL + 884736UL + 294912UL);
  __bf16* wfc2   = (__bf16*)(ws + 539492352UL + 884736UL + 294912UL + 1179648UL);
  __bf16* h2buf  = hbuf;     // dead after QKV GEMM
  __bf16* hidden = qkvbuf;   // qkv+ao regions == exactly M*1536*2 bytes
  float*  out    = (float*)d_out;

  // Weight conversions (tiny)
  cvt_kernel<<<(1152 * 384 + 255) / 256, 256, 0, stream>>>(qkv_w, wqkv, 1152 * 384);
  cvt_kernel<<<(384 * 384 + 255) / 256, 256, 0, stream>>>(proj_w, wproj, 384 * 384);
  cvt_kernel<<<(1536 * 384 + 255) / 256, 256, 0, stream>>>(fc1_w, wfc1, 1536 * 384);
  cvt_kernel<<<(384 * 1536 + 255) / 256, 256, 0, stream>>>(fc2_w, wfc2, 384 * 1536);

  // LN1 + shift + window partition
  ln_kernel<<<MTOK, 384, 0, stream>>>(x, g1, b1, hbuf, 1);

  // QKV projection (M x 1152 = M x 384 @ 1152 x 384^T)
  gemm_kernel<0><<<dim3(MTOK / BM, QKVDIM / BN), 256, 0, stream>>>(
      hbuf, wqkv, qkv_b, qkvbuf, nullptr, nullptr, QKVDIM, CDIM);

  // Windowed attention
  attn_kernel<<<dim3(NWIN, NHEAD), 128, 0, stream>>>(qkvbuf, relb, aobuf);

  // Output projection + window-reverse + unshift + residual -> xres
  gemm_kernel<1><<<dim3(MTOK / BM, CDIM / BN), 256, 0, stream>>>(
      aobuf, wproj, proj_b, nullptr, xres, x, CDIM, CDIM);

  // LN2
  ln_kernel<<<MTOK, 384, 0, stream>>>(xres, g2, b2, h2buf, 0);

  // FC1 + exact GELU
  gemm_kernel<2><<<dim3(MTOK / BM, HIDDIM / BN), 256, 0, stream>>>(
      h2buf, wfc1, fc1_b, hidden, nullptr, nullptr, HIDDIM, CDIM);

  // FC2 + residual -> final output
  gemm_kernel<3><<<dim3(MTOK / BM, CDIM / BN), 256, 0, stream>>>(
      hidden, wfc2, fc2_b, nullptr, out, xres, CDIM, HIDDIM);
}
